// OfflineLayer_70557722739213
// MI455X (gfx1250) — matrix operand order repaired and checked by reading, not compiled, not run on hardware
//
#include <hip/hip_runtime.h>

// ---------------------------------------------------------------------------
// ST-GCN block, MI455X (gfx1250, wave32, WMMA bf16 16x16x32, async-LDS DMA)
//
//   K0: repack conv_w -> Wbf[co][k=p*256+ci] (bf16)
//   K1: xg = graphAgg(x) per p (WMMA), fused 9-tap temporal sum -> xt bf16
//       layout xt[p][n][ci][m][32(wpad)]
//   K3: z[co,(m,w)] = sum_k Wbf * xt  (WMMA, K=768), double-buffered K-loop:
//       As staged with global_load_async_to_lds_b128 (ASYNCcnt),
//       Bs double-buffered through registers; fused bias, LayerNorm (shuffle
//       reduction), ReLU, residual, ReLU -> out
// ---------------------------------------------------------------------------

typedef __attribute__((ext_vector_type(16))) __bf16 v16bf;
typedef __attribute__((ext_vector_type(8)))  float  v8f;
typedef unsigned int u32x4 __attribute__((ext_vector_type(4)));
typedef __attribute__((address_space(3))) const unsigned short lds_cus;

union FragBF { v16bf v; u32x4 q[2]; };
union Q2H    { u32x4 q; unsigned short h[8]; };

#define N_   16
#define CIN  256
#define L_   300
#define V_   25
#define P_   3
#define COUT 256
#define KTOT (P_ * CIN)      // 768
#define WPAD 32
#define ASSZ (256 * 40)      // halves per As buffer
#define BSSZ (128 * 40)      // halves per Bs buffer

__device__ __forceinline__ unsigned short f2bf(float f) {
  unsigned int u = __builtin_bit_cast(unsigned int, f);
  u += 0x7FFFu + ((u >> 16) & 1u);            // round-to-nearest-even
  return (unsigned short)(u >> 16);
}

__device__ __forceinline__ v8f wmma_bf16(v16bf a, v16bf b, v8f c) {
  return __builtin_amdgcn_wmma_f32_16x16x32_bf16(false, a, false, b,
                                                 (short)0, c, false, false);
}

// byte offset of a __shared__ location within the workgroup LDS segment
__device__ __forceinline__ unsigned lds_off(const unsigned short* p) {
  return (unsigned)(unsigned long long)(lds_cus*)p;
}

// async DMA: 16 bytes global -> LDS, tracked by ASYNCcnt
__device__ __forceinline__ void async_b128(unsigned lds_byte_off,
                                           const unsigned short* gp) {
  asm volatile("global_load_async_to_lds_b128 %0, %1, off"
               :: "v"(lds_byte_off), "v"(gp) : "memory");
}
__device__ __forceinline__ void wait_async0() {
  asm volatile("s_wait_asynccnt 0x0" ::: "memory");
}

// ---------------------------------------------------------------------------
// K0: Wbf[co][p*256+ci] = bf16(conv_w[(p*COUT+co)*CIN + ci])
// ---------------------------------------------------------------------------
__global__ void k0_wprep(const float* __restrict__ conv_w,
                         unsigned short* __restrict__ Wbf) {
  int idx = blockIdx.x * 256 + threadIdx.x;        // 0 .. 256*768-1
  int co = idx / KTOT;
  int k  = idx - co * KTOT;
  int p  = k >> 8;
  int ci = k & 255;
  Wbf[idx] = f2bf(conv_w[((p << 8) + co) * CIN + ci]);
}

// ---------------------------------------------------------------------------
// K1: per (n, c, l-block of 64): graph aggregation via WMMA + temporal sum.
//   grid (5, 256, 16), block 160 (5 waves; one 16-row WMMA tile per wave)
// ---------------------------------------------------------------------------
__global__ __launch_bounds__(160) void k1_spatial(
    const float* __restrict__ x, const float* __restrict__ A,
    const float* __restrict__ edge, unsigned short* __restrict__ xt) {
  __shared__ __align__(16) unsigned short xstage[80 * 32];      // [row][vpad]
  __shared__ __align__(16) unsigned short awlds[P_ * 32 * 40];  // [p][w][vpad40]
  __shared__ float ygbuf[P_ * 80 * 33];                         // [p][row][wpad33]

  const int tid = threadIdx.x;
  const int n = blockIdx.z, c = blockIdx.y, lb = blockIdx.x * 64;

  // stage Aw^T (B matrix: lane = column w, contiguous v), zero-padded
  for (int idx = tid; idx < P_ * 32 * 40; idx += 160) {
    int p = idx / 1280, r = idx - p * 1280;
    int w = r / 40, v = r - w * 40;
    unsigned short h = 0;
    if (w < V_ && v < V_) {
      int a = (p * V_ + v) * V_ + w;
      h = f2bf(A[a] * edge[a]);
    }
    awlds[idx] = h;
  }
  // stage x rows l in [lb-8, lb+71], zero halo / pad
  const float* xb = x + (size_t)(n * CIN + c) * (L_ * V_);
  for (int idx = tid; idx < 80 * 32; idx += 160) {
    int i = idx >> 5, v = idx & 31, l = lb - 8 + i;
    unsigned short h = 0;
    if (v < V_ && l >= 0 && l < L_) h = f2bf(xb[l * V_ + v]);
    xstage[idx] = h;
  }
  __syncthreads();

  const int wv = tid >> 5, lane = tid & 31, l15 = lane & 15, hi = lane >> 4;

  // B fragments: Aw^T for 3 partitions x 2 column groups
  FragBF bfr[P_][2];
#pragma unroll
  for (int p = 0; p < P_; ++p)
#pragma unroll
    for (int g = 0; g < 2; ++g) {
      const unsigned short* pb = &awlds[(p * 32 + g * 16 + l15) * 40 + hi * 8];
      bfr[p][g].q[0] = *(const u32x4*)pb;
      bfr[p][g].q[1] = *(const u32x4*)(pb + 16);
    }
  // A fragment: this wave's 16 l-rows
  {
    const unsigned short* pa = &xstage[(wv * 16 + l15) * 32 + hi * 8];
    FragBF a;
    a.q[0] = *(const u32x4*)pa;
    a.q[1] = *(const u32x4*)(pa + 16);
#pragma unroll
    for (int p = 0; p < P_; ++p)
#pragma unroll
      for (int g = 0; g < 2; ++g) {
        v8f acc = {};
        acc = wmma_bf16(a.v, bfr[p][g].v, acc);
#pragma unroll
        for (int j = 0; j < 8; ++j)
          ygbuf[(p * 80 + wv * 16 + hi * 8 + j) * 33 + g * 16 + l15] = acc[j];
      }
  }
  __syncthreads();

  // temporal 9-tap causal sum over rows, write bf16 to xt[p][n][c][m][32]
  for (int idx = tid; idx < P_ * 64 * 32; idx += 160) {
    int p = idx >> 11, rem = idx & 2047;
    int mloc = rem >> 5, w = rem & 31;
    int m = lb + mloc;
    if (m < L_) {
      const float* yg = &ygbuf[(p * 80 + mloc) * 33 + w];
      float s = 0.f;
#pragma unroll
      for (int d = 0; d < 9; ++d) s += yg[d * 33];
      size_t o = (((size_t)((p * N_ + n) * CIN + c) * L_ + m) << 5) + w;
      xt[o] = f2bf(s);
    }
  }
}

// ---------------------------------------------------------------------------
// K3: out = relu(relu(LN(W*xt + bias_eff)) + x)
//   grid (75, 16), block 512 (16 waves: 4 co-waves x 4 m-waves)
//   tile: 256 co x 128 s (4 frames x 32 wpad), double-buffered K-loop 24 x 32
// ---------------------------------------------------------------------------
__global__ __launch_bounds__(512) void k3_gemm_ln(
    const unsigned short* __restrict__ Wbf, const unsigned short* __restrict__ xt,
    const float* __restrict__ xin, const float* __restrict__ A,
    const float* __restrict__ edge, const float* __restrict__ conv_b,
    const float* __restrict__ gamma, const float* __restrict__ beta,
    float* __restrict__ out) {
  __shared__ __align__(16) unsigned short As[2 * ASSZ];  // [buf][co][kpad40]
  __shared__ __align__(16) unsigned short Bs[2 * BSSZ];  // [buf][s][kpad40]
  __shared__ float redS[16], redQ[16];
  __shared__ float Sp[P_ * 32];   // colsum of Aw per partition

  const int tid = threadIdx.x;
  const int mblk = blockIdx.x, n = blockIdx.y;
  const int wv = tid >> 5, lane = tid & 31, l15 = lane & 15, hi = lane >> 4;
  const int co_w = wv >> 2, s_w = wv & 3;
  const int m = mblk * 4 + s_w;

  // prologue: Aw column sums (for analytic bias propagation)
  for (int idx = tid; idx < P_ * 32; idx += 512) {
    int p = idx >> 5, w = idx & 31;
    float s = 0.f;
    if (w < V_)
      for (int v = 0; v < V_; ++v) {
        int a = (p * V_ + v) * V_ + w;
        s += A[a] * edge[a];
      }
    Sp[idx] = s;
  }

  // per-thread staging coordinates (fixed across K-steps)
  const int co_st = tid >> 1, koff_st = (tid & 1) << 4;     // As: 16 halves each
  const unsigned aoff0 = lds_off(&As[co_st * 40 + koff_st]);
  const int kB = tid >> 4, s8 = (tid & 15) << 3;            // Bs: 8 halves each

  // ---- pipeline prologue: tile 0 into buffer 0 ----
  {
    const unsigned short* g = Wbf + co_st * KTOT + koff_st;
    async_b128(aoff0, g);
    async_b128(aoff0 + 16, g + 8);
    const unsigned short* gb =
        xt + ((size_t)(n * CIN + kB) * L_ + mblk * 4) * WPAD + s8;  // kt=0: p=0,ci=kB
    Q2H u0; u0.q = *(const u32x4*)gb;
#pragma unroll
    for (int j = 0; j < 8; ++j) Bs[(s8 + j) * 40 + kB] = u0.h[j];
    wait_async0();
  }

  v8f acc[4][2] = {};

  for (int kt = 0; kt < 24; ++kt) {
    __syncthreads();  // buf[kt&1] ready; all reads of buf[(kt+1)&1] retired
    const int cur = kt & 1, nxt = cur ^ 1;
    const bool pre = (kt < 23);

    Q2H u;
    if (pre) {  // issue next tile's traffic; latency hides under the WMMAs
      const int kn = kt + 1;
      const unsigned short* g = Wbf + co_st * KTOT + (kn << 5) + koff_st;
      const unsigned ao = aoff0 + (unsigned)nxt * (ASSZ * 2);
      async_b128(ao, g);
      async_b128(ao + 16, g + 8);
      const int p = kn >> 3, ci = ((kn & 7) << 5) + kB;
      const unsigned short* gb =
          xt + ((size_t)((p * N_ + n) * CIN + ci) * L_ + mblk * 4) * WPAD + s8;
      u.q = *(const u32x4*)gb;
    }

    const unsigned short* Abase = As + cur * ASSZ;
    const unsigned short* Bbase = Bs + cur * BSSZ;
    FragBF a[4], b[2];
#pragma unroll
    for (int ct = 0; ct < 4; ++ct) {
      const unsigned short* pa = Abase + (co_w * 64 + ct * 16 + l15) * 40 + hi * 8;
      a[ct].q[0] = *(const u32x4*)pa;
      a[ct].q[1] = *(const u32x4*)(pa + 16);
    }
#pragma unroll
    for (int g = 0; g < 2; ++g) {
      const unsigned short* pb = Bbase + (s_w * 32 + g * 16 + l15) * 40 + hi * 8;
      b[g].q[0] = *(const u32x4*)pb;
      b[g].q[1] = *(const u32x4*)(pb + 16);
    }
#pragma unroll
    for (int ct = 0; ct < 4; ++ct)
#pragma unroll
      for (int g = 0; g < 2; ++g)
        acc[ct][g] = wmma_bf16(a[ct].v, b[g].v, acc[ct][g]);

    if (pre) {  // transpose-store next Bs tile into the idle buffer
      unsigned short* Bd = Bs + nxt * BSSZ;
#pragma unroll
      for (int j = 0; j < 8; ++j) Bd[(s8 + j) * 40 + kB] = u.h[j];
    }
    wait_async0();  // this wave's async As writes landed before next barrier
  }
  __syncthreads();

  // epilogue: analytic bias (conv_b through agg/temporal), LN statistics
  const float cnt = (float)((m + 1 < 9) ? m + 1 : 9);
  float psum = 0.f, psq = 0.f;
#pragma unroll
  for (int ct = 0; ct < 4; ++ct)
#pragma unroll
    for (int g = 0; g < 2; ++g)
#pragma unroll
      for (int j = 0; j < 8; ++j) {
        int co = co_w * 64 + ct * 16 + hi * 8 + j;
        int w = g * 16 + l15;
        if (w < V_) {
          float z = acc[ct][g][j];
          z += cnt * (conv_b[co] * Sp[w] + conv_b[256 + co] * Sp[32 + w] +
                      conv_b[512 + co] * Sp[64 + w]);
          acc[ct][g][j] = z;
          psum += z;
          psq += z * z;
        }
      }
  // wave32 shuffle reduction, then one partial per wave
#pragma unroll
  for (int off = 16; off > 0; off >>= 1) {
    psum += __shfl_xor(psum, off, 32);
    psq  += __shfl_xor(psq,  off, 32);
  }
  if (lane == 0) { redS[wv] = psum; redQ[wv] = psq; }
  __syncthreads();
  // waves wv = co_w*4 + s_w share frame m with co_w = 0..3
  const float inv = 1.0f / (float)(COUT * V_);
  const float fs = redS[s_w] + redS[4 + s_w] + redS[8 + s_w] + redS[12 + s_w];
  const float fq = redQ[s_w] + redQ[4 + s_w] + redQ[8 + s_w] + redQ[12 + s_w];
  const float mu = fs * inv;
  const float var = fq * inv - mu * mu;
  const float rs = rsqrtf(var + 1e-5f);

#pragma unroll
  for (int ct = 0; ct < 4; ++ct)
#pragma unroll
    for (int g = 0; g < 2; ++g)
#pragma unroll
      for (int j = 0; j < 8; ++j) {
        int co = co_w * 64 + ct * 16 + hi * 8 + j;
        int w = g * 16 + l15;
        if (w < V_) {
          float z = acc[ct][g][j];
          float y = (z - mu) * rs * gamma[co * V_ + w] + beta[co * V_ + w];
          y = fmaxf(y, 0.f);
          size_t oi = ((size_t)(n * COUT + co) * L_ + m) * V_ + w;
          y += xin[oi];
          out[oi] = fmaxf(y, 0.f);
        }
      }
}

// ---------------------------------------------------------------------------
extern "C" void kernel_launch(void* const* d_in, const int* in_sizes, int n_in,
                              void* d_out, int out_size, void* d_ws, size_t ws_size,
                              hipStream_t stream) {
  const float* x      = (const float*)d_in[0];
  const float* A      = (const float*)d_in[1];
  const float* conv_w = (const float*)d_in[2];
  const float* conv_b = (const float*)d_in[3];
  const float* gamma  = (const float*)d_in[4];
  const float* beta   = (const float*)d_in[5];
  const float* edge   = (const float*)d_in[6];
  float* out = (float*)d_out;

  unsigned short* Wbf = (unsigned short*)d_ws;                       // 384 KiB
  unsigned short* xt  = (unsigned short*)((char*)d_ws + 393216);     // ~225 MiB

  k0_wprep<<<KTOT, 256, 0, stream>>>(conv_w, Wbf);
  k1_spatial<<<dim3(5, CIN, N_), 160, 0, stream>>>(x, A, edge, xt);
  k3_gemm_ln<<<dim3(L_ / 4, N_), 512, 0, stream>>>(Wbf, xt, x, A, edge,
                                                   conv_b, gamma, beta, out);
}
